// Router_5454608466016
// MI455X (gfx1250) — matrix operand order repaired and checked by reading, not compiled, not run on hardware
//
#include <hip/hip_runtime.h>
#include <hip/hip_bf16.h>

typedef __attribute__((ext_vector_type(16))) __bf16 v16bf;
typedef __attribute__((ext_vector_type(8)))  float  v8f;
typedef __attribute__((ext_vector_type(4)))  float  v4f;

#define NUM_GROUPS   8
#define GROUP_SIZE   4096
#define D_MODEL      1024
#define NUM_EXPERTS  64
#define TOTAL_TOKENS (NUM_GROUPS * GROUP_SIZE)   // 32768
#define BAL_COEF     0.04f
#define Z_COEF       0.001f

// ---------------------------------------------------------------------------
// K0: zero the outputs (masks/weights/loss) and the loss accumulators in ws.
// ---------------------------------------------------------------------------
__global__ void zero_kernel(float* __restrict__ out, int n_out, float* __restrict__ acc) {
    int i = blockIdx.x * blockDim.x + threadIdx.x;
    if (i < n_out) out[i] = 0.0f;
    if (i < 2)     acc[i] = 0.0f;
}

// ---------------------------------------------------------------------------
// K1: router GEMM.  logits[t][e] = sum_k x[t][k] * W[k][e]
// bf16 WMMA 16x16x32, fp32 accumulate. W staged transposed in LDS as bf16.
// One wave per 16-token M-tile; each wave owns all 4 expert N-tiles.
// ---------------------------------------------------------------------------
__global__ void gemm_kernel(const float* __restrict__ x,
                            const float* __restrict__ W,
                            float* __restrict__ logits) {
    extern __shared__ __bf16 Wt[];   // [NUM_EXPERTS][D_MODEL] = 128 KB

    const int tid = threadIdx.x;
    // Stage W (k-major in memory) transposed -> Wt[n][k], coalesced global reads.
    for (int i = tid; i < NUM_EXPERTS * D_MODEL; i += blockDim.x) {
        int k = i >> 6;        // i / 64
        int n = i & 63;        // i % 64
        Wt[n * D_MODEL + k] = (__bf16)W[i];
    }
    __syncthreads();

    const int wave  = tid >> 5;
    const int lane  = tid & 31;
    const int half  = lane >> 4;     // 0: lanes 0-15, 1: lanes 16-31
    const int l16   = lane & 15;
    const int tile_m = blockIdx.x * 8 + wave;      // 2048 M-tiles total
    const int row    = tile_m * 16 + l16;          // A-matrix: M = lane%16
    const float* xr  = x + (size_t)row * D_MODEL;

    v8f acc0 = {}, acc1 = {}, acc2 = {}, acc3 = {};

    for (int kk = 0; kk < D_MODEL; kk += 32) {
        // Prefetch next K-slab of x into caches.
        if (kk + 32 < D_MODEL)
            __builtin_prefetch(xr + kk + 32 + 8 * half, 0, 0);

        // A fragment: per lane, K = [kk+16h, +15] contiguous (matches B),
        // converted to bf16.
        v4f c0 = *(const v4f*)(xr + kk + 16 * half);
        v4f c1 = *(const v4f*)(xr + kk + 16 * half + 4);
        v4f c2 = *(const v4f*)(xr + kk + 16 * half + 8);
        v4f c3 = *(const v4f*)(xr + kk + 16 * half + 12);
        v16bf a;
#pragma unroll
        for (int i = 0; i < 4; ++i) {
            a[i]      = (__bf16)c0[i];
            a[4 + i]  = (__bf16)c1[i];
            a[8 + i]  = (__bf16)c2[i];
            a[12 + i] = (__bf16)c3[i];
        }

        // B fragments (same contiguous k order as A): lane holds N = lane%16,
        // K = [kk + 16*half, +15] contiguous in Wt -> ds_load_b128 pairs.
        const __bf16* wb = Wt + kk + 16 * half;
        v16bf b0 = *(const v16bf*)(wb + (size_t)(l16 +  0) * D_MODEL);
        v16bf b1 = *(const v16bf*)(wb + (size_t)(l16 + 16) * D_MODEL);
        v16bf b2 = *(const v16bf*)(wb + (size_t)(l16 + 32) * D_MODEL);
        v16bf b3 = *(const v16bf*)(wb + (size_t)(l16 + 48) * D_MODEL);

        acc0 = __builtin_amdgcn_wmma_f32_16x16x32_bf16(false, a, false, b0, (short)0, acc0, false, false);
        acc1 = __builtin_amdgcn_wmma_f32_16x16x32_bf16(false, a, false, b1, (short)0, acc1, false, false);
        acc2 = __builtin_amdgcn_wmma_f32_16x16x32_bf16(false, a, false, b2, (short)0, acc2, false, false);
        acc3 = __builtin_amdgcn_wmma_f32_16x16x32_bf16(false, a, false, b3, (short)0, acc3, false, false);
    }

    // C/D layout: VGPR r, lanes 0-15 -> M=r, lanes 16-31 -> M=r+8; N = lane%16.
#pragma unroll
    for (int r = 0; r < 8; ++r) {
        int m = tile_m * 16 + r + 8 * half;
        float* lr = logits + (size_t)m * NUM_EXPERTS + l16;
        lr[0]  = acc0[r];
        lr[16] = acc1[r];
        lr[32] = acc2[r];
        lr[48] = acc3[r];
    }
}

// ---------------------------------------------------------------------------
// K2: softmax over experts per token; write probs TRANSPOSED to P[e][t];
// accumulate z-loss (logsumexp^2) via one atomic per token.
// One wave32 per token (2 experts per lane).
// ---------------------------------------------------------------------------
__global__ void softmax_kernel(const float* __restrict__ logits,
                               float* __restrict__ P,
                               float* __restrict__ acc) {
    const int wave = threadIdx.x >> 5;
    const int lane = threadIdx.x & 31;
    const int t = blockIdx.x * (blockDim.x >> 5) + wave;

    const float* lr = logits + (size_t)t * NUM_EXPERTS + lane * 2;
    float lx = lr[0], ly = lr[1];

    float m = fmaxf(lx, ly);
#pragma unroll
    for (int off = 16; off; off >>= 1) m = fmaxf(m, __shfl_xor(m, off, 32));

    float ex = __expf(lx - m), ey = __expf(ly - m);
    float s = ex + ey;
#pragma unroll
    for (int off = 16; off; off >>= 1) s += __shfl_xor(s, off, 32);

    float inv = 1.0f / s;
    P[(size_t)(lane * 2 + 0) * TOTAL_TOKENS + t] = ex * inv;
    P[(size_t)(lane * 2 + 1) * TOTAL_TOKENS + t] = ey * inv;

    if (lane == 0) {
        float z = m + __logf(s);            // logsumexp
        atomicAdd(acc + 0, z * z);
    }
}

// ---------------------------------------------------------------------------
// K3: per-expert top-K selection via 11-bit histogram radix-select.
// Probs are positive -> raw float bits are order-preserving; key = bits>>21.
// Pass 1: histogram + row sum (-> balance loss). Pass 2: scatter selected.
// One workgroup per expert.
// ---------------------------------------------------------------------------
__global__ void select_kernel(const float* __restrict__ P,
                              const int* __restrict__ capPtr,
                              float* __restrict__ masks,
                              float* __restrict__ weights,
                              float* __restrict__ acc) {
    __shared__ unsigned hist[2048];
    __shared__ float    ssum;
    __shared__ unsigned sc_bin, sc_rem, sc_cnt;

    const int e   = blockIdx.x;
    const int tid = threadIdx.x;
    const float* Pe = P + (size_t)e * TOTAL_TOKENS;

    for (int i = tid; i < 2048; i += blockDim.x) hist[i] = 0u;
    if (tid == 0) { ssum = 0.0f; sc_cnt = 0u; sc_bin = 0u; sc_rem = 0u; }
    __syncthreads();

    float ls = 0.0f;
    for (int t = tid; t < TOTAL_TOKENS; t += blockDim.x) {
        float p = Pe[t];
        ls += p;
        unsigned bin = __float_as_uint(p) >> 21;
        atomicAdd(&hist[bin], 1u);
    }
    atomicAdd(&ssum, ls);
    __syncthreads();

    if (tid == 0) {
        unsigned K = (unsigned)capPtr[0];
        unsigned cum = 0;
        for (int b = 2047; b >= 0; --b) {
            unsigned c = hist[b];
            if (cum + c >= K) { sc_bin = (unsigned)b; sc_rem = K - cum; break; }
            cum += c;
        }
        float usage = ssum / ((float)TOTAL_TOKENS + 1e-5f);
        atomicAdd(acc + 1, usage * usage);
    }
    __syncthreads();

    const unsigned B = sc_bin, rem = sc_rem;
    for (int t = tid; t < TOTAL_TOKENS; t += blockDim.x) {
        float p = Pe[t];
        unsigned bin = __float_as_uint(p) >> 21;
        bool sel = bin > B;
        if (bin == B) {
            unsigned slot = atomicAdd(&sc_cnt, 1u);
            sel = slot < rem;                    // tie-break within threshold bin
        }
        if (sel) {
            masks  [(size_t)e * TOTAL_TOKENS + t] = 1.0f;
            weights[(size_t)e * TOTAL_TOKENS + t] = p;
        }
    }
}

// ---------------------------------------------------------------------------
// K4: combine accumulators into the final scalar loss.
// ---------------------------------------------------------------------------
__global__ void finalize_kernel(const float* __restrict__ acc,
                                float* __restrict__ loss_out) {
    float bal = ((float)NUM_EXPERTS * acc[1] - 1.0f) * BAL_COEF;
    float zl  = Z_COEF * acc[0] / (float)TOTAL_TOKENS;
    *loss_out = bal + zl;
}

// ---------------------------------------------------------------------------
extern "C" void kernel_launch(void* const* d_in, const int* in_sizes, int n_in,
                              void* d_out, int out_size, void* d_ws, size_t ws_size,
                              hipStream_t stream) {
    const float* x   = (const float*)d_in[0];
    const float* W   = (const float*)d_in[1];
    const int*   cap = (const int*)d_in[2];

    float* out     = (float*)d_out;
    float* masks   = out;                                        // (E, G, S) as 0/1 floats
    float* weights = out + (size_t)NUM_EXPERTS * TOTAL_TOKENS;   // (E, G, S)
    float* loss    = out + (size_t)2 * NUM_EXPERTS * TOTAL_TOKENS;

    float* ws     = (float*)d_ws;
    float* P      = ws;                                          // [E][T] probs  (8 MB)
    float* logits = ws + (size_t)NUM_EXPERTS * TOTAL_TOKENS;     // [T][E] logits (8 MB)
    float* acc    = ws + (size_t)2 * NUM_EXPERTS * TOTAL_TOKENS; // [0]=sum z^2, [1]=sum usage^2

    const int n_out = 2 * NUM_EXPERTS * TOTAL_TOKENS + 1;
    zero_kernel<<<(n_out + 255) / 256, 256, 0, stream>>>(out, n_out, acc);

    // 2048 M-tiles, 8 waves/block -> 256 blocks; 128 KB dynamic LDS for Wt.
    gemm_kernel<<<TOTAL_TOKENS / 16 / 8, 256,
                  NUM_EXPERTS * D_MODEL * sizeof(__bf16), stream>>>(x, W, logits);

    // one wave per token, 8 waves/block -> 4096 blocks
    softmax_kernel<<<TOTAL_TOKENS / 8, 256, 0, stream>>>(logits, P, acc);

    // one workgroup per expert
    select_kernel<<<NUM_EXPERTS, 256, 0, stream>>>(P, cap, masks, weights, acc);

    finalize_kernel<<<1, 1, 0, stream>>>(acc, loss);
}
